// AttentionModule_51230369906805
// MI455X (gfx1250) — compile-verified, matrix operand order checked by reading, not run on hardware
//
#include <hip/hip_runtime.h>
#include <hip/hip_bf16.h>

typedef __attribute__((ext_vector_type(16))) _Float16 v16h;
typedef __attribute__((ext_vector_type(8)))  float    v8f;
typedef __attribute__((ext_vector_type(4)))  float    v4f;

#define BB   32
#define SS   4096
#define DD   1024
#define MBLK 64       // rows per workgroup (128.25KB LDS -> 2 WGs per WGP)
#define NT   256      // threads per block (8 waves of 32)

// ---------------------------------------------------------------------------
// decb[b][e] = sum_d h[b,d] * Ua_w[e,d] + Ua_b[e] + Wa_b[e]
// ---------------------------------------------------------------------------
__global__ void decb_kernel(const float* __restrict__ h,
                            const float* __restrict__ Uaw,
                            const float* __restrict__ Uab,
                            const float* __restrict__ Wab,
                            float* __restrict__ decb) {
  int idx = blockIdx.x * blockDim.x + threadIdx.x;   // BB*DD total
  int b = idx >> 10;
  int e = idx & (DD - 1);
  const float4* hr = (const float4*)(h + (size_t)b * DD);
  const float4* wr = (const float4*)(Uaw + (size_t)e * DD);
  float s = 0.f;
#pragma unroll 4
  for (int d = 0; d < DD / 4; ++d) {
    float4 a = hr[d], w = wr[d];
    s += a.x * w.x + a.y * w.y + a.z * w.z + a.w * w.w;
  }
  decb[idx] = s + Uab[e] + Wab[e];
}

// ---------------------------------------------------------------------------
// Pack Wa_w (f32 [e][d]) into f16 WMMA B-fragments (ISA 7.12.2 layout):
//   lane l<16 : N = l,     K in {kc*32+0..7, kc*32+16..23}
//   lane l>=16: N = l-16,  K in {kc*32+8..15, kc*32+24..31}
// B[k][n] = Wa_w[nBase+n][k].  Slot id = (nc*32+kc)*32+lane, 16 halves each.
// ---------------------------------------------------------------------------
__global__ void pack_kernel(const float* __restrict__ Waw,
                            _Float16* __restrict__ Bp) {
  int slot = blockIdx.x * blockDim.x + threadIdx.x;  // 64*32*32 = 65536
  int lane = slot & 31;
  int kc   = (slot >> 5) & 31;
  int nc   = slot >> 10;
  int n    = (nc << 4) + (lane & 15);
  int koff = (kc << 5) + ((lane & 16) ? 8 : 0);
  const float* src = Waw + (size_t)n * DD + koff;
  v16h hv;
#pragma unroll
  for (int i = 0; i < 8; ++i) hv[i] = (_Float16)src[i];
#pragma unroll
  for (int i = 0; i < 8; ++i) hv[8 + i] = (_Float16)src[16 + i];
  *(v16h*)(Bp + (size_t)slot * 16) = hv;
}

// ---------------------------------------------------------------------------
// Fused GEMM + tanh + Va-dot.
// WG = 64 rows; 8 waves = 4 m-subtiles x 2 e-halves.  Each wave: 8 acc tiles
// (64 VGPRs), 4 e-sweeps x 32 K-steps x 8 WMMAs.  A tile in LDS (f16
// fragments); B fragments loaded from global (shared addrs -> WGP$ hits).
// Raw scores written into d_out; softmax kernel follows.
// ---------------------------------------------------------------------------
__global__ void score_kernel(const float* __restrict__ enc,
                             const _Float16* __restrict__ Bp,
                             const float* __restrict__ decb,
                             const float* __restrict__ vaw,
                             const float* __restrict__ vab,
                             float* __restrict__ scores) {
  extern __shared__ char smem[];
  _Float16* Atile    = (_Float16*)smem;                        // MBLK*DD f16
  float*    scoreLds = (float*)(smem + (size_t)MBLK * DD * 2); // MBLK floats

  const int t = threadIdx.x;
  const size_t rowBase = (size_t)blockIdx.x * MBLK;
  const int b = (int)(rowBase >> 12);          // rowBase / 4096 (MBLK | S)
  const float* Abase = enc + rowBase * DD;

  if (t < MBLK) scoreLds[t] = 0.f;

  // ---- stage A: MBLK rows x 1024 f32 -> f16 fragments in LDS ----
  // slot = (msub, kc, lane); lane l<16: row msub*16+l, K {0..7,16..23};
  // lane l>=16: row msub*16+(l-16), K {8..15,24..31}.
  // Non-temporal: enc is streamed once; keep Bp/decb resident in L2.
#pragma unroll
  for (int i = 0; i < (MBLK * DD / 16) / NT; ++i) {            // 16 iters
    int slot = t + i * NT;
    int lane = slot & 31;
    int kc   = (slot >> 5) & 31;
    int msub = slot >> 10;
    int row  = (msub << 4) + (lane & 15);
    int koff = (kc << 5) + ((lane & 16) ? 8 : 0);
    const float* src = Abase + (size_t)row * DD + koff;
    v4f f0 = __builtin_nontemporal_load((const v4f*)(src));
    v4f f1 = __builtin_nontemporal_load((const v4f*)(src + 4));
    v4f g0 = __builtin_nontemporal_load((const v4f*)(src + 16));
    v4f g1 = __builtin_nontemporal_load((const v4f*)(src + 20));
    v16h hv;
    hv[0]=(_Float16)f0.x;  hv[1]=(_Float16)f0.y;  hv[2]=(_Float16)f0.z;  hv[3]=(_Float16)f0.w;
    hv[4]=(_Float16)f1.x;  hv[5]=(_Float16)f1.y;  hv[6]=(_Float16)f1.z;  hv[7]=(_Float16)f1.w;
    hv[8]=(_Float16)g0.x;  hv[9]=(_Float16)g0.y;  hv[10]=(_Float16)g0.z; hv[11]=(_Float16)g0.w;
    hv[12]=(_Float16)g1.x; hv[13]=(_Float16)g1.y; hv[14]=(_Float16)g1.z; hv[15]=(_Float16)g1.w;
    *(v16h*)(Atile + (size_t)slot * 16) = hv;
  }
  __syncthreads();

  const int lane  = t & 31;
  const int wave  = t >> 5;
  const int msub  = wave >> 1;                 // m-subtile id (0..3)
  const int ehalf = wave & 1;                  // e-half (0..1)
  const v16h* Av = (const v16h*)Atile;
  const v16h* Bv = (const v16h*)Bp;

  float partial[8] = {0.f,0.f,0.f,0.f,0.f,0.f,0.f,0.f};
  const v8f vzero = {0.f,0.f,0.f,0.f,0.f,0.f,0.f,0.f};

  for (int sw = 0; sw < 4; ++sw) {             // 4 sweeps x 128 e per wave
    const int sweep = (ehalf << 2) + sw;       // e range: sweep*128 .. +127
    v8f acc[8];
#pragma unroll
    for (int n = 0; n < 8; ++n) acc[n] = vzero;

    for (int kc = 0; kc < 32; ++kc) {
      v16h a = Av[((msub << 5) + kc) * 32 + lane];
#pragma unroll
      for (int n = 0; n < 8; ++n) {
        int nc = (sweep << 3) + n;
        v16h bf = Bv[(size_t)(((nc << 5) + kc) * 32 + lane)];
        acc[n] = __builtin_amdgcn_wmma_f32_16x16x32_f16(
            false, a, false, bf, (short)0, acc[n], false, false);
      }
    }

    // fused epilogue: tanh(acc + dec) * Va -> per-row partials
    // C layout: lane<16 -> N=lane, VGPR j -> M=j; lane>=16 -> N=lane-16, M=8+j
#pragma unroll
    for (int n = 0; n < 8; ++n) {
      int e = (sweep << 7) + (n << 4) + (lane & 15);
      float add = decb[(b << 10) + e];
      float vw  = vaw[e];
#pragma unroll
      for (int j = 0; j < 8; ++j)
        partial[j] += tanhf(acc[n][j] + add) * vw;
    }
  }

  // reduce partials across each 16-lane half (e dimension)
#pragma unroll
  for (int off = 1; off < 16; off <<= 1) {
#pragma unroll
    for (int j = 0; j < 8; ++j)
      partial[j] += __shfl_xor(partial[j], off, 32);
  }
  if ((lane & 15) == 0) {
    int mb = (msub << 4) + ((lane >> 4) << 3); // rows msub*16 + {0..7|8..15}
#pragma unroll
    for (int j = 0; j < 8; ++j)
      atomicAdd(&scoreLds[mb + j], partial[j]);
  }
  __syncthreads();
  if (t < MBLK) scores[rowBase + t] = scoreLds[t] + vab[0];
}

// ---------------------------------------------------------------------------
// Masked softmax over S=4096, one block per batch, in-place on d_out.
// ---------------------------------------------------------------------------
__global__ void softmax_kernel(float* __restrict__ io,
                               const unsigned char* __restrict__ mask) {
  int bidx = blockIdx.x;
  int t = threadIdx.x;
  int lane = t & 31, wave = t >> 5;
  __shared__ float red[8];
  size_t base = (size_t)bidx * SS;

  float loc[16];
  float mx = -INFINITY;
#pragma unroll
  for (int i = 0; i < 16; ++i) {
    int s = t + i * NT;
    float v = io[base + s];
    if (!mask[base + s]) v = -INFINITY;
    loc[i] = v;
    mx = fmaxf(mx, v);
  }
#pragma unroll
  for (int off = 1; off < 32; off <<= 1) mx = fmaxf(mx, __shfl_xor(mx, off, 32));
  if (lane == 0) red[wave] = mx;
  __syncthreads();
  float m = red[0];
#pragma unroll
  for (int w = 1; w < 8; ++w) m = fmaxf(m, red[w]);

  float sum = 0.f;
#pragma unroll
  for (int i = 0; i < 16; ++i) {
    float e2 = __expf(loc[i] - m);   // exp(-inf) = 0 for masked lanes
    loc[i] = e2;
    sum += e2;
  }
#pragma unroll
  for (int off = 1; off < 32; off <<= 1) sum += __shfl_xor(sum, off, 32);
  __syncthreads();                   // all reads of red(max) complete
  if (lane == 0) red[wave] = sum;
  __syncthreads();
  float tot = 0.f;
#pragma unroll
  for (int w = 0; w < 8; ++w) tot += red[w];
  float inv = 1.0f / tot;
#pragma unroll
  for (int i = 0; i < 16; ++i) io[base + t + i * NT] = loc[i] * inv;
}

// ---------------------------------------------------------------------------
extern "C" void kernel_launch(void* const* d_in, const int* in_sizes, int n_in,
                              void* d_out, int out_size, void* d_ws, size_t ws_size,
                              hipStream_t stream) {
  const float* enc          = (const float*)d_in[0];  // [B,S,D]
  const float* h            = (const float*)d_in[1];  // [1,B,D]
  const unsigned char* mask = (const unsigned char*)d_in[2]; // [B,S] bool
  const float* Waw          = (const float*)d_in[3];  // [D,D]
  const float* Wab          = (const float*)d_in[4];  // [D]
  const float* Uaw          = (const float*)d_in[5];  // [D,D]
  const float* Uab          = (const float*)d_in[6];  // [D]
  const float* Vaw          = (const float*)d_in[7];  // [1,D]
  const float* Vab          = (const float*)d_in[8];  // [1]
  float* out = (float*)d_out;                         // [B,1,S]

  float*    decb = (float*)d_ws;                                  // 128 KB
  _Float16* Bp   = (_Float16*)((char*)d_ws + (size_t)BB * DD * sizeof(float)); // 2 MB

  decb_kernel<<<(BB * DD) / NT, NT, 0, stream>>>(h, Uaw, Uab, Wab, decb);
  pack_kernel<<<(64 * 32 * 32) / NT, NT, 0, stream>>>(Waw, Bp);

  size_t smem = (size_t)MBLK * DD * sizeof(_Float16) + MBLK * sizeof(float);
  score_kernel<<<(BB * SS) / MBLK, NT, smem, stream>>>(enc, Bp, decb, Vaw, Vab, out);

  softmax_kernel<<<BB, NT, 0, stream>>>(out, mask);
}
